// LeggedTransformerDistill_39625368272984
// MI455X (gfx1250) — matrix operand order repaired and checked by reading, not compiled, not run on hardware
//
#include <hip/hip_runtime.h>
#include <cstdint>

// Problem constants (from reference)
#define BB 8
#define TT 512
#define CC 512
#define LL 4
#define NH 8
#define HD 64
#define SDIM 48
#define ADIM 12
#define SS 1024              // 2*T interleaved sequence
#define MM (BB*SS)           // 8192 token rows
#define FF (4*CC)            // 2048 MLP hidden

typedef __attribute__((ext_vector_type(16))) __bf16 bf16x16;
typedef __attribute__((ext_vector_type(8)))  __bf16 bf16x8;
typedef __attribute__((ext_vector_type(8)))  float  f32x8;

// ---------- small helpers ----------
__device__ __forceinline__ __bf16 f2bf(float f){
    union { float f; uint32_t u; } c; c.f = f;
    uint32_t r = c.u + 0x7FFFu + ((c.u >> 16) & 1u);   // round-to-nearest-even
    union { uint16_t u; __bf16 b; } o; o.u = (uint16_t)(r >> 16);
    return o.b;
}

__device__ __forceinline__ bf16x8 ldb8(const __bf16* p){
    return *reinterpret_cast<const bf16x8*>(p);
}

__device__ __forceinline__ bf16x16 mkfrag(bf16x8 lo, bf16x8 hi){
    bf16x16 r;
#pragma unroll
    for (int i = 0; i < 8; ++i){ r[i] = lo[i]; r[i+8] = hi[i]; }
    return r;
}

__device__ __forceinline__ float gelu_exact(float x){
    return 0.5f * x * (1.0f + erff(x * 0.70710678118654752f));
}

// Async copy of 16 bytes global -> LDS (GVS mode: uniform 64-bit base in SGPRs,
// per-lane 32-bit byte offset in VADDR, LDS byte address in VDST). Tracked by ASYNCcnt.
__device__ __forceinline__ void async_g2lds_b128(uint32_t lds_addr, uint32_t goff, const void* sbase){
    asm volatile("global_load_async_to_lds_b128 %0, %1, %2"
                 :: "v"(lds_addr), "v"(goff), "s"(sbase) : "memory");
}
template<int N>
__device__ __forceinline__ void wait_asynccnt(){
    asm volatile("s_wait_asynccnt %0" :: "i"(N) : "memory");
}
__device__ __forceinline__ uint32_t lds_addr_of(const void* p){
    return (uint32_t)(uintptr_t)p;     // LDS aperture: low 32 bits = LDS byte offset
}

// ---------- fp32 -> bf16 weight conversion ----------
__global__ void cvt_f32_bf16_kernel(const float* __restrict__ s, __bf16* __restrict__ d, long n){
    long i = (long)blockIdx.x * blockDim.x + threadIdx.x;
    long st = (long)gridDim.x * blockDim.x;
    for (; i < n; i += st) d[i] = f2bf(s[i]);
}

// ---------- fused embedding + interleave + LayerNorm ----------
__global__ __launch_bounds__(256) void embed_ln_kernel(
    const float* __restrict__ states, const float* __restrict__ actions,
    const float* __restrict__ We_s, const float* __restrict__ be_s,
    const float* __restrict__ We_a, const float* __restrict__ be_a,
    const float* __restrict__ pe, const float* __restrict__ g, const float* __restrict__ bta,
    float* __restrict__ h, __bf16* __restrict__ hb)
{
    __shared__ float inb[64];
    __shared__ float red[256];
    int row = blockIdx.x;                 // 0..8191
    int b = row >> 10, s = row & 1023, t = s >> 1;
    bool is_state = ((s & 1) == 0);
    int tid = threadIdx.x;
    int K = is_state ? SDIM : ADIM;
    const float* src = is_state ? (states + (size_t)(b*TT + t) * SDIM)
                                : (actions + (size_t)(b*TT + t) * ADIM);
    if (tid < K) inb[tid] = src[tid];
    __syncthreads();
    const float* Wm = is_state ? We_s : We_a;
    const float* bm = is_state ? be_s : be_a;
    float v[2];
#pragma unroll
    for (int p = 0; p < 2; ++p){
        int c = tid + p*256;
        float a = bm[c] + pe[t*CC + c];
        const float* wr = Wm + (size_t)c * K;
        for (int k = 0; k < K; ++k) a += inb[k] * wr[k];
        v[p] = a;
    }
    // LayerNorm over CC=512
    red[tid] = v[0] + v[1]; __syncthreads();
    for (int st2 = 128; st2 > 0; st2 >>= 1){ if (tid < st2) red[tid] += red[tid+st2]; __syncthreads(); }
    float mean = red[0] * (1.0f/CC); __syncthreads();
    float d0 = v[0]-mean, d1 = v[1]-mean;
    red[tid] = d0*d0 + d1*d1; __syncthreads();
    for (int st2 = 128; st2 > 0; st2 >>= 1){ if (tid < st2) red[tid] += red[tid+st2]; __syncthreads(); }
    float rstd = rsqrtf(red[0] * (1.0f/CC) + 1e-5f);
#pragma unroll
    for (int p = 0; p < 2; ++p){
        int c = tid + p*256;
        float o = (v[p]-mean)*rstd*g[c] + bta[c];
        h [(size_t)row*CC + c] = o;
        hb[(size_t)row*CC + c] = f2bf(o);
    }
}

// ---------- LayerNorm (fp32 in -> fp32 + bf16 out) ----------
__global__ __launch_bounds__(256) void ln_kernel(
    const float* __restrict__ x, const float* __restrict__ g, const float* __restrict__ bta,
    float* __restrict__ h, __bf16* __restrict__ hb)
{
    __shared__ float red[256];
    int row = blockIdx.x, tid = threadIdx.x;
    float v0 = x[(size_t)row*CC + tid], v1 = x[(size_t)row*CC + tid + 256];
    red[tid] = v0 + v1; __syncthreads();
    for (int st = 128; st > 0; st >>= 1){ if (tid < st) red[tid] += red[tid+st]; __syncthreads(); }
    float mean = red[0] * (1.0f/CC); __syncthreads();
    float d0 = v0-mean, d1 = v1-mean;
    red[tid] = d0*d0 + d1*d1; __syncthreads();
    for (int st = 128; st > 0; st >>= 1){ if (tid < st) red[tid] += red[tid+st]; __syncthreads(); }
    float rstd = rsqrtf(red[0] * (1.0f/CC) + 1e-5f);
    float o0 = d0*rstd*g[tid]     + bta[tid];
    float o1 = d1*rstd*g[tid+256] + bta[tid+256];
    h [(size_t)row*CC + tid]       = o0;
    h [(size_t)row*CC + tid + 256] = o1;
    hb[(size_t)row*CC + tid]       = f2bf(o0);
    hb[(size_t)row*CC + tid + 256] = f2bf(o1);
}

// ---------- WMMA GEMM: Y[M,N] = X[M,K] @ W[N,K]^T + bias ----------
// B tiles (64 cols x 64 K, 8 KB) are double-buffered in LDS via async copies:
// tile k+1 streams in (ASYNCcnt) while all 8 waves run 8 WMMAs on tile k.
// MODE 0: bf16 store; MODE 1: fp32 store with residual add; MODE 2: GELU + bf16 store
template<int MODE>
__global__ __launch_bounds__(256) void wmma_gemm_kernel(
    const __bf16* __restrict__ X, const __bf16* __restrict__ W,
    const float* __restrict__ bias, const float* __restrict__ res,
    float* __restrict__ outF, __bf16* __restrict__ outB,
    int M, int Nout, int K)
{
    __shared__ __align__(16) __bf16 btile[2][64*64];   // [buf][col][k], 2 x 8 KB

    int tid  = threadIdx.x;
    int lane = tid & 31;
    int wave = tid >> 5;
    int half = lane >> 4;
    int rowbase = blockIdx.y * 128 + wave * 16;
    int colbase = blockIdx.x * 64;
    int r = rowbase + (lane & 15);

    // async-stage mapping: thread -> one 32-byte chunk of the 64x64 tile (2 B128s)
    int brow = tid >> 2;                 // 0..63 : W row (output column)
    int bchk = tid & 3;                  // 0..3  : 32-byte chunk in the 128-byte K-row
    const __bf16* wbase = W + (size_t)colbase * K;                     // uniform (SGPR)
    uint32_t boff0 = (uint32_t)(((size_t)brow * K + bchk * 16) * 2);   // per-lane byte offset
    uint32_t blds[2];
    blds[0] = lds_addr_of(&btile[0][brow*64 + bchk*16]);
    blds[1] = lds_addr_of(&btile[1][brow*64 + bchk*16]);

    f32x8 acc[4];
#pragma unroll
    for (int j = 0; j < 4; ++j)
#pragma unroll
        for (int e = 0; e < 8; ++e) acc[j][e] = 0.0f;

    // prologue: stage tile 0
    async_g2lds_b128(blds[0],      boff0,      wbase);
    async_g2lds_b128(blds[0] + 32, boff0 + 32, wbase);

    for (int kk = 0; kk < K; kk += 64){
        int cur = (kk >> 6) & 1;
        if (kk + 64 < K){
            // prefetch next tile into the alternate buffer, keep it in flight
            uint32_t goff = boff0 + (uint32_t)((kk + 64) * 2);
            async_g2lds_b128(blds[cur ^ 1],      goff,      wbase);
            async_g2lds_b128(blds[cur ^ 1] + 32, goff + 32, wbase);
            wait_asynccnt<2>();       // drain current tile, leave prefetch pending
        } else {
            wait_asynccnt<0>();
        }

        // A fragments stream from global (per-wave rows) while copies fly
        const __bf16* ap0 = X + (size_t)r * K + kk + half * 16;
        bf16x16 a0 = mkfrag(ldb8(ap0), ldb8(ap0 + 8));
        bf16x16 a1 = mkfrag(ldb8(ap0 + 32), ldb8(ap0 + 40));
        if (kk + 64 < K) __builtin_prefetch(ap0 + 64, 0, 0);   // global_prefetch_b8

        __syncthreads();              // publish all waves' LDS writes

        const __bf16* bt = btile[cur];
#pragma unroll
        for (int j = 0; j < 4; ++j){
            const __bf16* bp = bt + (j*16 + (lane & 15)) * 64 + half * 16;
            bf16x16 b0 = mkfrag(ldb8(bp),      ldb8(bp + 8));      // ds_load_b128
            bf16x16 b1 = mkfrag(ldb8(bp + 32), ldb8(bp + 40));
            acc[j] = __builtin_amdgcn_wmma_f32_16x16x32_bf16(
                false, a0, false, b0, (short)0, acc[j], false, false);
            acc[j] = __builtin_amdgcn_wmma_f32_16x16x32_bf16(
                false, a1, false, b1, (short)0, acc[j], false, false);
        }
        __syncthreads();              // tile consumed before restage
    }

#pragma unroll
    for (int j = 0; j < 4; ++j){
        int col = colbase + j*16 + (lane & 15);
        float bv = bias[col];
#pragma unroll
        for (int e = 0; e < 8; ++e){
            int row = rowbase + e + 8*half;
            size_t idx = (size_t)row * Nout + col;
            float v = acc[j][e] + bv;
            if (MODE == 0) outB[idx] = f2bf(v);
            else if (MODE == 1) outF[idx] = v + res[idx];
            else outB[idx] = f2bf(gelu_exact(v));
        }
    }
}

// ---------- causal multi-head attention (per b,h,16-query-row tile) ----------
__global__ __launch_bounds__(256) void attn_kernel(
    const __bf16* __restrict__ qb, const __bf16* __restrict__ kb,
    const __bf16* __restrict__ vb, __bf16* __restrict__ attb)
{
    extern __shared__ __align__(16) char smem_raw[];
    float*  sc    = reinterpret_cast<float*>(smem_raw);                 // 16 x 1024 fp32 (64 KB)
    __bf16* P     = reinterpret_cast<__bf16*>(smem_raw + 16*SS*4);      // 16 x 1024 bf16 (32 KB)
    __bf16* vt0   = reinterpret_cast<__bf16*>(smem_raw + 16*SS*4 + 16*SS*2);          // 32x64 bf16
    __bf16* vt1   = reinterpret_cast<__bf16*>(smem_raw + 16*SS*4 + 16*SS*2 + 32*64*2);// 32x64 bf16

    int qt = blockIdx.x;              // 0..63 (query tile of 16 rows)
    int hh = blockIdx.y;              // head
    int b  = blockIdx.z;              // batch
    int tid  = threadIdx.x;
    int lane = tid & 31;
    int wave = tid >> 5;
    int half = lane >> 4;
    int hc = hh * HD;
    int rowg0 = b * SS + qt * 16;
    const float scale = 0.125f;       // 1/sqrt(64)

    // Phase 1: scores = Q @ K^T (each wave owns 8 of 64 key tiles)
    int qrow = rowg0 + (lane & 15);
    for (int it = 0; it < 8; ++it){
        int jt = it * 8 + wave;
        int keybase = jt * 16;
        f32x8 acc;
#pragma unroll
        for (int e = 0; e < 8; ++e) acc[e] = 0.0f;
#pragma unroll
        for (int ks = 0; ks < 2; ++ks){
            int kk = ks * 32;
            const __bf16* ap = qb + (size_t)qrow * CC + hc + kk + half * 16;
            bf16x16 a = mkfrag(ldb8(ap), ldb8(ap + 8));
            const __bf16* bp = kb + (size_t)(b*SS + keybase + (lane & 15)) * CC + hc + kk + half * 16;
            bf16x16 bfr = mkfrag(ldb8(bp), ldb8(bp + 8));
            acc = __builtin_amdgcn_wmma_f32_16x16x32_bf16(
                false, a, false, bfr, (short)0, acc, false, false);
        }
        int key = keybase + (lane & 15);
#pragma unroll
        for (int e = 0; e < 8; ++e){
            int qloc = e + 8*half;
            int qseq = qt*16 + qloc;
            float v = acc[e] * scale;
            if (key > qseq) v = -1e30f;          // causal mask
            sc[qloc * SS + key] = v;
        }
    }
    __syncthreads();

    // Phase 2: softmax; 16 lanes per row, reductions stay inside a wave32
    {
        int row = tid >> 4;     // 0..15
        int sub = tid & 15;
        float m = -1e30f;
        for (int j = 0; j < 64; ++j) m = fmaxf(m, sc[row*SS + sub + j*16]);
#pragma unroll
        for (int off = 1; off < 16; off <<= 1) m = fmaxf(m, __shfl_xor(m, off, 32));
        float s = 0.0f;
        for (int j = 0; j < 64; ++j){
            int idx = row*SS + sub + j*16;
            float e = __expf(sc[idx] - m);
            sc[idx] = e;
            s += e;
        }
#pragma unroll
        for (int off = 1; off < 16; off <<= 1) s += __shfl_xor(s, off, 32);
        float inv = 1.0f / s;
        for (int j = 0; j < 64; ++j){
            int idx = row*SS + sub + j*16;
            P[idx] = f2bf(sc[idx] * inv);
        }
    }
    __syncthreads();

    // Phase 3: out = P @ V. All 8 waves double-buffer the 32x64 V tile through
    // LDS via async copies; waves 0..3 each compute a 16-wide slice of head dim.
    int vrow = tid >> 3;     // 0..31 : key row within tile
    int vchk = tid & 7;      // 0..7  : 16-byte chunk of the 128-byte V row
    const __bf16* vbase = vb + (size_t)(b*SS) * CC + hc;              // uniform (SGPR)
    uint32_t voff0 = (uint32_t)(((size_t)vrow * CC + vchk * 8) * 2);  // per-lane byte offset
    uint32_t vlds[2];
    vlds[0] = lds_addr_of(vt0 + vrow*64 + vchk*8);
    vlds[1] = lds_addr_of(vt1 + vrow*64 + vchk*8);
    int dloc = wave*16 + (lane & 15);   // local head-dim column for compute waves

    f32x8 acc;
#pragma unroll
    for (int e = 0; e < 8; ++e) acc[e] = 0.0f;

    async_g2lds_b128(vlds[0], voff0, vbase);   // prologue: stage tile 0

    for (int kk = 0; kk < SS; kk += 32){
        int cur = (kk >> 5) & 1;
        if (kk + 32 < SS){
            async_g2lds_b128(vlds[cur ^ 1], voff0 + (uint32_t)((size_t)(kk + 32) * CC * 2), vbase);
            wait_asynccnt<1>();       // drain current tile, leave prefetch pending
        } else {
            wait_asynccnt<0>();
        }
        __syncthreads();
        if (wave < 4){
            const __bf16* ap = P + (size_t)(lane & 15) * SS + kk + half * 16;
            bf16x16 a = mkfrag(ldb8(ap), ldb8(ap + 8));
            const __bf16* vt = cur ? vt1 : vt0;
            bf16x16 bfr;
#pragma unroll
            for (int i = 0; i < 16; ++i)
                bfr[i] = vt[(half*16 + i) * 64 + dloc];
            acc = __builtin_amdgcn_wmma_f32_16x16x32_bf16(
                false, a, false, bfr, (short)0, acc, false, false);
        }
        __syncthreads();
    }

    if (wave < 4){
#pragma unroll
        for (int e = 0; e < 8; ++e){
            int row = rowg0 + e + 8*half;
            attb[(size_t)row * CC + hc + dloc] = f2bf(acc[e]);
        }
    }
}

// ---------- output heads (narrow GEMMs, plain fp32) ----------
__global__ __launch_bounds__(64) void heads_kernel(
    const float* __restrict__ h,
    const float* __restrict__ Wps, const float* __restrict__ bps,
    const float* __restrict__ Wpa, const float* __restrict__ bpa,
    float* __restrict__ out)
{
    int bt = blockIdx.x;              // 0..B*T-1
    int b = bt / TT, t = bt % TT;
    int j = threadIdx.x;
    if (j < SDIM){
        const float* hr = h + (size_t)(b*SS + 2*t + 1) * CC;   // action-token position
        const float* wr = Wps + (size_t)j * CC;
        float a = bps[j];
        for (int k = 0; k < CC; ++k) a += hr[k] * wr[k];
        out[(size_t)bt * SDIM + j] = a;
    } else if (j < SDIM + ADIM){
        int jj = j - SDIM;
        const float* hr = h + (size_t)(b*SS + 2*t) * CC;       // state-token position
        const float* wr = Wpa + (size_t)jj * CC;
        float a = bpa[jj];
        for (int k = 0; k < CC; ++k) a += hr[k] * wr[k];
        out[(size_t)(BB*TT)*SDIM + (size_t)bt * ADIM + jj] = a;
    }
}

// ---------- orchestration ----------
extern "C" void kernel_launch(void* const* d_in, const int* in_sizes, int n_in,
                              void* d_out, int out_size, void* d_ws, size_t ws_size,
                              hipStream_t stream)
{
    const float* states  = (const float*)d_in[0];
    const float* actions = (const float*)d_in[1];
    const float* We_s = (const float*)d_in[2];  const float* be_s = (const float*)d_in[3];
    const float* We_a = (const float*)d_in[4];  const float* be_a = (const float*)d_in[5];
    const float* pe   = (const float*)d_in[6];
    const float* g_emb = (const float*)d_in[7]; const float* b_emb = (const float*)d_in[8];
    const float* Wq = (const float*)d_in[9];   const float* bq = (const float*)d_in[10];
    const float* Wk = (const float*)d_in[11];  const float* bk = (const float*)d_in[12];
    const float* Wv = (const float*)d_in[13];  const float* bv = (const float*)d_in[14];
    const float* Wo = (const float*)d_in[15];  const float* bo = (const float*)d_in[16];
    const float* W1 = (const float*)d_in[17];  const float* b1 = (const float*)d_in[18];
    const float* W2 = (const float*)d_in[19];  const float* b2 = (const float*)d_in[20];
    const float* g1 = (const float*)d_in[21];  const float* bt1 = (const float*)d_in[22];
    const float* g2 = (const float*)d_in[23];  const float* bt2 = (const float*)d_in[24];
    const float* Wps = (const float*)d_in[25]; const float* bps = (const float*)d_in[26];
    const float* Wpa = (const float*)d_in[27]; const float* bpa = (const float*)d_in[28];
    float* out = (float*)d_out;

    // workspace carve-up
    char* ws = (char*)d_ws;
    size_t off = 0;
    auto alloc = [&](size_t bytes)->char*{
        char* p = ws + off; off = (off + bytes + 255) & ~(size_t)255; return p;
    };
    float*  h    = (float*) alloc((size_t)MM*CC*4);
    __bf16* hb   = (__bf16*)alloc((size_t)MM*CC*2);
    float*  t0   = (float*) alloc((size_t)MM*CC*4);
    __bf16* qbuf = (__bf16*)alloc((size_t)MM*CC*2);
    __bf16* kbuf = (__bf16*)alloc((size_t)MM*CC*2);
    __bf16* vbuf = (__bf16*)alloc((size_t)MM*CC*2);
    __bf16* abuf = (__bf16*)alloc((size_t)MM*CC*2);
    __bf16* ubuf = (__bf16*)alloc((size_t)MM*FF*2);
    __bf16* WqB  = (__bf16*)alloc((size_t)LL*CC*CC*2);
    __bf16* WkB  = (__bf16*)alloc((size_t)LL*CC*CC*2);
    __bf16* WvB  = (__bf16*)alloc((size_t)LL*CC*CC*2);
    __bf16* WoB  = (__bf16*)alloc((size_t)LL*CC*CC*2);
    __bf16* W1B  = (__bf16*)alloc((size_t)LL*FF*CC*2);
    __bf16* W2B  = (__bf16*)alloc((size_t)LL*CC*FF*2);
    (void)ws_size; (void)in_sizes; (void)n_in; (void)out_size;

    // 1) weights -> bf16
    long nqkv = (long)LL*CC*CC, nmlp = (long)LL*FF*CC;
    cvt_f32_bf16_kernel<<<2048, 256, 0, stream>>>(Wq, WqB, nqkv);
    cvt_f32_bf16_kernel<<<2048, 256, 0, stream>>>(Wk, WkB, nqkv);
    cvt_f32_bf16_kernel<<<2048, 256, 0, stream>>>(Wv, WvB, nqkv);
    cvt_f32_bf16_kernel<<<2048, 256, 0, stream>>>(Wo, WoB, nqkv);
    cvt_f32_bf16_kernel<<<2048, 256, 0, stream>>>(W1, W1B, nmlp);
    cvt_f32_bf16_kernel<<<2048, 256, 0, stream>>>(W2, W2B, nmlp);

    // 2) embed + interleave + LN
    embed_ln_kernel<<<MM, 256, 0, stream>>>(states, actions, We_s, be_s, We_a, be_a,
                                            pe, g_emb, b_emb, h, hb);

    dim3 gQKV(CC/64, MM/128);      // 8 x 64
    dim3 gFC1(FF/64, MM/128);      // 32 x 64
    dim3 gAtt(SS/16, NH, BB);      // 64 x 8 x 8
    size_t attSmem = (size_t)16*SS*4 + (size_t)16*SS*2 + (size_t)2*32*64*2;   // 104 KB

    for (int i = 0; i < LL; ++i){
        const __bf16* wq = WqB + (size_t)i*CC*CC;
        const __bf16* wk = WkB + (size_t)i*CC*CC;
        const __bf16* wv = WvB + (size_t)i*CC*CC;
        const __bf16* wo = WoB + (size_t)i*CC*CC;
        const __bf16* w1 = W1B + (size_t)i*FF*CC;
        const __bf16* w2 = W2B + (size_t)i*CC*FF;

        // Q,K,V projections (bf16 out)
        wmma_gemm_kernel<0><<<gQKV, 256, 0, stream>>>(hb, wq, bq + i*CC, nullptr, nullptr, qbuf, MM, CC, CC);
        wmma_gemm_kernel<0><<<gQKV, 256, 0, stream>>>(hb, wk, bk + i*CC, nullptr, nullptr, kbuf, MM, CC, CC);
        wmma_gemm_kernel<0><<<gQKV, 256, 0, stream>>>(hb, wv, bv + i*CC, nullptr, nullptr, vbuf, MM, CC, CC);

        // causal attention
        attn_kernel<<<gAtt, 256, attSmem, stream>>>(qbuf, kbuf, vbuf, abuf);

        // output projection + residual -> t0 (fp32), then LN1
        wmma_gemm_kernel<1><<<gQKV, 256, 0, stream>>>(abuf, wo, bo + i*CC, h, t0, nullptr, MM, CC, CC);
        ln_kernel<<<MM, 256, 0, stream>>>(t0, g1 + i*CC, bt1 + i*CC, h, hb);

        // MLP: fc1 + GELU (bf16), fc2 + residual (fp32), LN2
        wmma_gemm_kernel<2><<<gFC1, 256, 0, stream>>>(hb, w1, b1 + i*FF, nullptr, nullptr, ubuf, MM, FF, CC);
        wmma_gemm_kernel<1><<<gQKV, 256, 0, stream>>>(ubuf, w2, b2 + i*CC, h, t0, nullptr, MM, CC, FF);
        ln_kernel<<<MM, 256, 0, stream>>>(t0, g2 + i*CC, bt2 + i*CC, h, hb);
    }

    // output heads
    heads_kernel<<<BB*TT, 64, 0, stream>>>(h, Wps, bps, Wpa, bpa, out);
}
